// WaterMDDynamicBoxNet_14499809591856
// MI455X (gfx1250) — matrix operand order repaired and checked by reading, not compiled, not run on hardware
//
#include <hip/hip_runtime.h>
#include <hip/hip_bf16.h>

// ---------------------------------------------------------------------------
// GNN block for MI455X (gfx1250): WMMA f32_16x16x32_f16 GEMMs, fused edge and
// node pipelines, fp32 scatter-add aggregation. 64 rows/block with B-fragment
// register reuse across two M-tiles to halve L2 weight traffic.
// ---------------------------------------------------------------------------

typedef __attribute__((ext_vector_type(16))) _Float16 v16h;
typedef __attribute__((ext_vector_type(8)))  _Float16 v8h;
typedef __attribute__((ext_vector_type(8)))  float    v8f;

#define ASTRIDE 136   // padded LDS row stride in halves (272B -> bank rotation)
#define D128 128

union V16U { v16h v; v8h h[2]; };

__device__ __forceinline__ v8f wmma_f16(v16h a, v16h b, v8f c) {
    // D = A(16x32 f16) * B(32x16 f16) + C(16x16 f32)
    return __builtin_amdgcn_wmma_f32_16x16x32_f16(false, a, false, b,
                                                  (short)0, c, false, false);
}

// A fragment from a padded LDS activation tile (row-major, ASTRIDE halves/row).
__device__ __forceinline__ v16h frag_a_lds(const _Float16* base, int row_base,
                                           int kb, int lane) {
    int row = row_base + (lane & 15);
    int off = (kb << 5) + ((lane >> 4) << 3);   // K-halves offset
    V16U u;
    u.h[0] = *(const v8h*)(base + row * ASTRIDE + off);        // K lo chunk
    u.h[1] = *(const v8h*)(base + row * ASTRIDE + off + 16);   // K hi chunk
    return u.v;
}

// A fragment from a global fp16 row (row ptr per lane pair, stride-free).
__device__ __forceinline__ v16h frag_a_row(const _Float16* rowptr, int kb, int lane) {
    int off = (kb << 5) + ((lane >> 4) << 3);
    V16U u;
    u.h[0] = *(const v8h*)(rowptr + off);
    u.h[1] = *(const v8h*)(rowptr + off + 16);
    return u.v;
}

// B fragment from pre-packed weights: tile (kb,nb) -> 512 halves, lane-major.
__device__ __forceinline__ v16h frag_b(const _Float16* wmat, int kb, int nb, int lane) {
    return *(const v16h*)(wmat + ((((kb << 3) + nb) << 9) + (lane << 4)));
}

__device__ __forceinline__ v8f bcast8(float x) {
    v8f a;
#pragma unroll
    for (int i = 0; i < 8; ++i) a[i] = x;
    return a;
}

__device__ __forceinline__ float silu_f(float x) {
    return x / (1.0f + __expf(-x));
}

// Store silu(acc) into padded LDS tile at (rows mt*16 + (lane>>4)*8 + v, col).
__device__ __forceinline__ void store_silu(_Float16* T, int mt, int col, int lane, v8f acc) {
    int r0 = (mt << 4) + ((lane >> 4) << 3);
#pragma unroll
    for (int v = 0; v < 8; ++v)
        T[(r0 + v) * ASTRIDE + col] = (_Float16)silu_f(acc[v]);
}

__device__ __forceinline__ void atomicAddF(float* p, float v) {
    (void)__hip_atomic_fetch_add(p, v, __ATOMIC_RELAXED, __HIP_MEMORY_SCOPE_AGENT);
}

// Dual-M-tile GEMM: B fragments loaded once, used by both M-tiles.
__device__ __forceinline__ void gemm2_rowA(const _Float16* ar0, const _Float16* ar1,
                                           const _Float16* wp, int nb0, int nb1, int lane,
                                           v8f& c00, v8f& c01, v8f& c10, v8f& c11) {
#pragma unroll
    for (int kb = 0; kb < 4; ++kb) {
        v16h b0 = frag_b(wp, kb, nb0, lane);
        v16h b1 = frag_b(wp, kb, nb1, lane);
        v16h a0 = frag_a_row(ar0, kb, lane);
        v16h a1 = frag_a_row(ar1, kb, lane);
        c00 = wmma_f16(a0, b0, c00);
        c01 = wmma_f16(a0, b1, c01);
        c10 = wmma_f16(a1, b0, c10);
        c11 = wmma_f16(a1, b1, c11);
    }
}

__device__ __forceinline__ void gemm2_ldsA(const _Float16* T, int mt0, int mt1,
                                           const _Float16* wp, int nb0, int nb1, int lane,
                                           v8f& c00, v8f& c01, v8f& c10, v8f& c11) {
#pragma unroll
    for (int kb = 0; kb < 4; ++kb) {
        v16h b0 = frag_b(wp, kb, nb0, lane);
        v16h b1 = frag_b(wp, kb, nb1, lane);
        v16h a0 = frag_a_lds(T, mt0 << 4, kb, lane);
        v16h a1 = frag_a_lds(T, mt1 << 4, kb, lane);
        c00 = wmma_f16(a0, b0, c00);
        c01 = wmma_f16(a0, b1, c01);
        c10 = wmma_f16(a1, b0, c10);
        c11 = wmma_f16(a1, b1, c11);
    }
}

// ---------------------------------------------------------------------------
// Prep kernels
// ---------------------------------------------------------------------------

// Pack 36 (= 4 layers x 9 kinds) 128x128 fp32 weight matrices into the WMMA
// B-fragment layout as fp16. kind: 0 ea_w1, 1 ea_w2, 2 src_w, 3 dst_w,
// 4 th_w1, 5 th_w2, 6 pd_w, 7 pe_w, 8 phi_w.
__global__ void __launch_bounds__(256)
pack_weights(const float* __restrict__ ea_w1, const float* __restrict__ ea_w2,
             const float* __restrict__ src_w, const float* __restrict__ dst_w,
             const float* __restrict__ th_w1, const float* __restrict__ th_w2,
             const float* __restrict__ pd_w,  const float* __restrict__ pe_w,
             const float* __restrict__ phi_w, _Float16* __restrict__ packed) {
    int idx = blockIdx.x * 256 + threadIdx.x;
    if (idx >= 36 * 16384) return;
    int mat    = idx >> 14;
    int layer  = mat / 9;
    int kind   = mat % 9;
    int r      = idx & 16383;
    int tile   = r >> 9;          // kb*8 + nb
    int within = r & 511;
    int lane   = within >> 4;
    int j      = within & 15;
    int kb = tile >> 3, nb = tile & 7;
    int k = (kb << 5) + ((lane >> 4) << 3) + (j & 7) + ((j >> 3) << 4);   // 0..127
    int n = (nb << 4) + (lane & 15);              // 0..127
    const float* W;
    switch (kind) {
        case 0: W = ea_w1; break; case 1: W = ea_w2; break;
        case 2: W = src_w; break; case 3: W = dst_w; break;
        case 4: W = th_w1; break; case 5: W = th_w2; break;
        case 6: W = pd_w;  break; case 7: W = pe_w;  break;
        default: W = phi_w; break;
    }
    packed[idx] = (_Float16)W[(size_t)layer * 16384 + k * D128 + n];
}

__global__ void __launch_bounds__(256)
convert_e(const float* __restrict__ e, _Float16* __restrict__ e16, size_t total) {
    for (size_t i = (size_t)blockIdx.x * 256 + threadIdx.x; i < total;
         i += (size_t)gridDim.x * 256)
        e16[i] = (_Float16)e[i];
}

// ---------------------------------------------------------------------------
// BatchNorm
// ---------------------------------------------------------------------------

__global__ void bn_zero(float* stats) {
    if (threadIdx.x < 256) stats[threadIdx.x] = 0.0f;
}

__global__ void __launch_bounds__(128)
bn_reduce(const float* __restrict__ h, float* __restrict__ stats, int Ntot) {
    int c = threadIdx.x;
    float s = 0.0f, s2 = 0.0f;
    for (int r = blockIdx.x; r < Ntot; r += gridDim.x) {
        float v = h[(size_t)r * D128 + c];
        s += v; s2 += v * v;
    }
    atomicAddF(&stats[c], s);
    atomicAddF(&stats[D128 + c], s2);
}

__global__ void __launch_bounds__(256)
bn_apply(const float* __restrict__ h, const float* __restrict__ stats,
         const float* __restrict__ gw, const float* __restrict__ gb,
         float* __restrict__ hn32, _Float16* __restrict__ hn16,
         float* __restrict__ agg, int total, float invN) {
    for (size_t i = (size_t)blockIdx.x * 256 + threadIdx.x; i < (size_t)total;
         i += (size_t)gridDim.x * 256) {
        int c = (int)(i & 127);
        float mu  = stats[c] * invN;
        float var = stats[D128 + c] * invN - mu * mu;
        float x = (h[i] - mu) * rsqrtf(var + 1e-5f) * gw[c] + gb[c];
        hn32[i] = x;
        hn16[i] = (_Float16)x;
        agg[i]  = 0.0f;
    }
}

// ---------------------------------------------------------------------------
// Fused edge pipeline: 64 edges / block, 8 waves; each wave owns two 16-row
// M-tiles x one 32-col N-pair so every B fragment feeds 2 WMMAs.
// ---------------------------------------------------------------------------

__global__ void __launch_bounds__(256)
edge_kernel(const _Float16* __restrict__ e16, const _Float16* __restrict__ hn16,
            const float* __restrict__ hn32, const int* __restrict__ src_idx,
            const int* __restrict__ dst_idx, float* __restrict__ agg,
            const _Float16* __restrict__ wl,
            const float* __restrict__ b_ea1, const float* __restrict__ b_ea2,
            const float* __restrict__ b_src, const float* __restrict__ b_dst,
            const float* __restrict__ b_th1, const float* __restrict__ b_th2,
            int Etot) {
    __shared__ __align__(16) _Float16 T0[64 * ASTRIDE];
    __shared__ __align__(16) _Float16 T1[64 * ASTRIDE];

    const int lane = threadIdx.x & 31;
    const int wave = threadIdx.x >> 5;
    const int mt0  = (wave >> 2) << 1;     // 0 or 2
    const int mt1  = mt0 + 1;              // 1 or 3
    const int nb0  = (wave & 3) << 1;      // N-block pair
    const int nb1  = nb0 + 1;
    const int ebase = blockIdx.x * 64;

    const int col0 = (nb0 << 4) + (lane & 15);
    const int col1 = (nb1 << 4) + (lane & 15);
    int ea0 = ebase + (mt0 << 4) + (lane & 15); if (ea0 >= Etot) ea0 = Etot - 1;
    int ea1 = ebase + (mt1 << 4) + (lane & 15); if (ea1 >= Etot) ea1 = Etot - 1;
    const int s0 = src_idx[ea0], s1 = src_idx[ea1];
    const int d0 = dst_idx[ea0], d1 = dst_idx[ea1];

    // Phase 1: T0 = silu(e @ ea_w1 + ea_b1)
    {
        v8f c00 = bcast8(b_ea1[col0]), c01 = bcast8(b_ea1[col1]);
        v8f c10 = c00, c11 = c01;
        gemm2_rowA(e16 + (size_t)ea0 * D128, e16 + (size_t)ea1 * D128,
                   wl + 0 * 16384, nb0, nb1, lane, c00, c01, c10, c11);
        store_silu(T0, mt0, col0, lane, c00); store_silu(T0, mt0, col1, lane, c01);
        store_silu(T0, mt1, col0, lane, c10); store_silu(T0, mt1, col1, lane, c11);
    }
    __syncthreads();

    // Phase 2: T1 = silu(T0 @ ea_w2 + hn[src] @ src_w + hn[dst] @ dst_w + biases)
    {
        v8f c00 = bcast8(b_ea2[col0] + b_src[col0] + b_dst[col0]);
        v8f c01 = bcast8(b_ea2[col1] + b_src[col1] + b_dst[col1]);
        v8f c10 = c00, c11 = c01;
        gemm2_ldsA(T0, mt0, mt1, wl + 1 * 16384, nb0, nb1, lane, c00, c01, c10, c11);
        gemm2_rowA(hn16 + (size_t)s0 * D128, hn16 + (size_t)s1 * D128,
                   wl + 2 * 16384, nb0, nb1, lane, c00, c01, c10, c11);
        gemm2_rowA(hn16 + (size_t)d0 * D128, hn16 + (size_t)d1 * D128,
                   wl + 3 * 16384, nb0, nb1, lane, c00, c01, c10, c11);
        store_silu(T1, mt0, col0, lane, c00); store_silu(T1, mt0, col1, lane, c01);
        store_silu(T1, mt1, col0, lane, c10); store_silu(T1, mt1, col1, lane, c11);
    }
    __syncthreads();

    // Phase 3: T0 = silu(T1 @ th_w1 + th_b1)
    {
        v8f c00 = bcast8(b_th1[col0]), c01 = bcast8(b_th1[col1]);
        v8f c10 = c00, c11 = c01;
        gemm2_ldsA(T1, mt0, mt1, wl + 4 * 16384, nb0, nb1, lane, c00, c01, c10, c11);
        store_silu(T0, mt0, col0, lane, c00); store_silu(T0, mt0, col1, lane, c01);
        store_silu(T0, mt1, col0, lane, c10); store_silu(T0, mt1, col1, lane, c11);
    }
    __syncthreads();

    // Phase 4: e_emb = T0 @ th_w2 + th_b2 ; scatter hn[src]*e_emb into agg[dst]
    {
        v8f c00 = bcast8(b_th2[col0]), c01 = bcast8(b_th2[col1]);
        v8f c10 = c00, c11 = c01;
        gemm2_ldsA(T0, mt0, mt1, wl + 5 * 16384, nb0, nb1, lane, c00, c01, c10, c11);
        int r0 = (lane >> 4) << 3;
#pragma unroll
        for (int v = 0; v < 8; ++v) {
            int e0 = ebase + (mt0 << 4) + r0 + v;
            if (e0 < Etot) {
                int s = src_idx[e0];
                int d = dst_idx[e0];
                atomicAddF(&agg[(size_t)d * D128 + col0],
                           hn32[(size_t)s * D128 + col0] * c00[v]);
                atomicAddF(&agg[(size_t)d * D128 + col1],
                           hn32[(size_t)s * D128 + col1] * c01[v]);
            }
            int e1 = ebase + (mt1 << 4) + r0 + v;
            if (e1 < Etot) {
                int s = src_idx[e1];
                int d = dst_idx[e1];
                atomicAddF(&agg[(size_t)d * D128 + col0],
                           hn32[(size_t)s * D128 + col0] * c10[v]);
                atomicAddF(&agg[(size_t)d * D128 + col1],
                           hn32[(size_t)s * D128 + col1] * c11[v]);
            }
        }
    }
}

// ---------------------------------------------------------------------------
// Fused node update: out = h + phi(silu(hn@pd_w + agg@pe_w + b)) @ phi_w + b
// 64 nodes / block, same dual-M-tile B reuse.
// ---------------------------------------------------------------------------

__global__ void __launch_bounds__(256)
node_kernel(const _Float16* __restrict__ hn16, const float* __restrict__ agg,
            const float* __restrict__ hsrc, float* __restrict__ hdst,
            const _Float16* __restrict__ wl,
            const float* __restrict__ b_pd, const float* __restrict__ b_pe,
            const float* __restrict__ b_phi, int Ntot) {
    __shared__ __align__(16) _Float16 AG[64 * ASTRIDE];
    __shared__ __align__(16) _Float16 T0[64 * ASTRIDE];

    const int lane = threadIdx.x & 31;
    const int wave = threadIdx.x >> 5;
    const int mt0  = (wave >> 2) << 1;
    const int mt1  = mt0 + 1;
    const int nb0  = (wave & 3) << 1;
    const int nb1  = nb0 + 1;
    const int nbase = blockIdx.x * 64;

    // Stage agg tile fp32 -> fp16 LDS
    for (int i = threadIdx.x; i < 64 * D128; i += 256) {
        int r = i >> 7, c = i & 127;
        int node = nbase + r;
        float v = (node < Ntot) ? agg[(size_t)node * D128 + c] : 0.0f;
        AG[r * ASTRIDE + c] = (_Float16)v;
    }
    __syncthreads();

    const int col0 = (nb0 << 4) + (lane & 15);
    const int col1 = (nb1 << 4) + (lane & 15);
    int na0 = nbase + (mt0 << 4) + (lane & 15); if (na0 >= Ntot) na0 = Ntot - 1;
    int na1 = nbase + (mt1 << 4) + (lane & 15); if (na1 >= Ntot) na1 = Ntot - 1;

    // Phase 1: T0 = silu(hn @ pd_w + agg @ pe_w + pd_b + pe_b)
    {
        v8f c00 = bcast8(b_pd[col0] + b_pe[col0]);
        v8f c01 = bcast8(b_pd[col1] + b_pe[col1]);
        v8f c10 = c00, c11 = c01;
        gemm2_rowA(hn16 + (size_t)na0 * D128, hn16 + (size_t)na1 * D128,
                   wl + 6 * 16384, nb0, nb1, lane, c00, c01, c10, c11);
        gemm2_ldsA(AG, mt0, mt1, wl + 7 * 16384, nb0, nb1, lane, c00, c01, c10, c11);
        store_silu(T0, mt0, col0, lane, c00); store_silu(T0, mt0, col1, lane, c01);
        store_silu(T0, mt1, col0, lane, c10); store_silu(T0, mt1, col1, lane, c11);
    }
    __syncthreads();

    // Phase 2: out = T0 @ phi_w + phi_b + h (residual)
    {
        v8f c00 = bcast8(b_phi[col0]), c01 = bcast8(b_phi[col1]);
        v8f c10 = c00, c11 = c01;
        gemm2_ldsA(T0, mt0, mt1, wl + 8 * 16384, nb0, nb1, lane, c00, c01, c10, c11);
        int r0 = (lane >> 4) << 3;
#pragma unroll
        for (int v = 0; v < 8; ++v) {
            int n0 = nbase + (mt0 << 4) + r0 + v;
            if (n0 < Ntot) {
                size_t i0 = (size_t)n0 * D128 + col0;
                size_t i1 = (size_t)n0 * D128 + col1;
                hdst[i0] = hsrc[i0] + c00[v];
                hdst[i1] = hsrc[i1] + c01[v];
            }
            int n1 = nbase + (mt1 << 4) + r0 + v;
            if (n1 < Ntot) {
                size_t i0 = (size_t)n1 * D128 + col0;
                size_t i1 = (size_t)n1 * D128 + col1;
                hdst[i0] = hsrc[i0] + c10[v];
                hdst[i1] = hsrc[i1] + c11[v];
            }
        }
    }
}

// ---------------------------------------------------------------------------
// Host launcher
// ---------------------------------------------------------------------------

extern "C" void kernel_launch(void* const* d_in, const int* in_sizes, int n_in,
                              void* d_out, int out_size, void* d_ws, size_t ws_size,
                              hipStream_t stream) {
    (void)in_sizes; (void)n_in; (void)out_size; (void)ws_size;

    const float* h_in  = (const float*)d_in[0];
    const float* e     = (const float*)d_in[1];
    const int*   src   = (const int*)d_in[2];
    const int*   dst   = (const int*)d_in[3];
    const float* bn_w  = (const float*)d_in[4];
    const float* bn_b  = (const float*)d_in[5];
    const float* ea_w1 = (const float*)d_in[6];
    const float* ea_b1 = (const float*)d_in[7];
    const float* ea_w2 = (const float*)d_in[8];
    const float* ea_b2 = (const float*)d_in[9];
    const float* src_w = (const float*)d_in[10];
    const float* src_b = (const float*)d_in[11];
    const float* dst_w = (const float*)d_in[12];
    const float* dst_b = (const float*)d_in[13];
    const float* th_w1 = (const float*)d_in[14];
    const float* th_b1 = (const float*)d_in[15];
    const float* th_w2 = (const float*)d_in[16];
    const float* th_b2 = (const float*)d_in[17];
    const float* pd_w  = (const float*)d_in[18];
    const float* pd_b  = (const float*)d_in[19];
    const float* pe_w  = (const float*)d_in[20];
    const float* pe_b  = (const float*)d_in[21];
    const float* phi_w = (const float*)d_in[22];
    const float* phi_b = (const float*)d_in[23];

    const int Nn = 50000, Ee = 400000, Ll = 4;

    char* w = (char*)d_ws;
    auto carve = [&](size_t bytes) -> void* {
        void* p = (void*)w;
        w += (bytes + 255) & ~(size_t)255;
        return p;
    };
    float*     stats  = (float*)carve(256 * sizeof(float));
    float*     hn32   = (float*)carve((size_t)Nn * D128 * sizeof(float));
    _Float16*  hn16   = (_Float16*)carve((size_t)Nn * D128 * sizeof(_Float16));
    float*     agg    = (float*)carve((size_t)Nn * D128 * sizeof(float));
    float*     h_ws   = (float*)carve((size_t)Nn * D128 * sizeof(float));
    _Float16*  e16    = (_Float16*)carve((size_t)Ee * D128 * sizeof(_Float16));
    _Float16*  packed = (_Float16*)carve((size_t)36 * 16384 * sizeof(_Float16));

    pack_weights<<<(36 * 16384 + 255) / 256, 256, 0, stream>>>(
        ea_w1, ea_w2, src_w, dst_w, th_w1, th_w2, pd_w, pe_w, phi_w, packed);
    convert_e<<<2048, 256, 0, stream>>>(e, e16, (size_t)Ee * D128);

    for (int l = 0; l < Ll; ++l) {
        const float* hsrc = (l == 0) ? h_in : h_ws;
        float* hdst = (l == Ll - 1) ? (float*)d_out : h_ws;

        bn_zero<<<1, 256, 0, stream>>>(stats);
        bn_reduce<<<512, 128, 0, stream>>>(hsrc, stats, Nn);
        bn_apply<<<1024, 256, 0, stream>>>(hsrc, stats, bn_w + l * D128,
                                           bn_b + l * D128, hn32, hn16, agg,
                                           Nn * D128, 1.0f / (float)Nn);

        const _Float16* wl = packed + (size_t)l * 9 * 16384;
        edge_kernel<<<(Ee + 63) / 64, 256, 0, stream>>>(
            e16, hn16, hn32, src, dst, agg, wl,
            ea_b1 + l * D128, ea_b2 + l * D128, src_b + l * D128,
            dst_b + l * D128, th_b1 + l * D128, th_b2 + l * D128, Ee);

        node_kernel<<<(Nn + 63) / 64, 256, 0, stream>>>(
            hn16, agg, hsrc, hdst, wl,
            pd_b + l * D128, pe_b + l * D128, phi_b + l * D128, Nn);
    }
}
